// CausalSelfAttention_6047313953561
// MI455X (gfx1250) — matrix operand order repaired and checked by reading, not compiled, not run on hardware
//
#include <hip/hip_runtime.h>
#include <math.h>

// ---------------- problem constants ----------------
constexpr int B_   = 4;
constexpr int T_   = 2048;
constexpr int C_   = 1024;   // EMB
constexpr int H_   = 16;
constexpr int D_   = 64;     // head dim
constexpr int N3_  = 3 * C_; // 3072
constexpr int M_   = B_ * T_;                      // 8192 rows
constexpr size_t BHTD = (size_t)B_ * H_ * T_ * D_; // 8388608

typedef __attribute__((ext_vector_type(16))) __bf16 v16bf;
typedef __attribute__((ext_vector_type(8)))  __bf16 v8bf;
typedef __attribute__((ext_vector_type(4)))  __bf16 v4bf;
typedef __attribute__((ext_vector_type(8)))  float  v8f;

#define WMMA_BF16(a, b, c) \
  __builtin_amdgcn_wmma_f32_16x16x32_bf16(false, (a), false, (b), (short)0, (c), false, false)

#define SHUF16(lo, hi) __builtin_shufflevector((lo), (hi), \
  0, 1, 2, 3, 4, 5, 6, 7, 8, 9, 10, 11, 12, 13, 14, 15)

__device__ __forceinline__ int imin(int a, int b) { return a < b ? a : b; }

// A-layout fragment from 32 consecutive bf16 (one K=32 chunk of a row):
// k = e + 16*hi
__device__ __forceinline__ v16bf load_a_bf16(const __bf16* ar, int hi) {
  const v8bf lo = *(const v8bf*)(ar + 16 * hi);
  const v8bf hh = *(const v8bf*)(ar + 16 * hi + 8);
  return SHUF16(lo, hh);
}

// ======================================================================
// Kernel 0a: x (f32, row-major) -> xb (bf16, row-major). 4 elems/thread.
// ======================================================================
__global__ __launch_bounds__(256) void conv_x_kernel(
    const float* __restrict__ x, __bf16* __restrict__ xb) {
  const size_t base = ((size_t)blockIdx.x * 256 + threadIdx.x) * 4;
  const float4 f = *(const float4*)(x + base);
  v4bf p;
  p[0] = (__bf16)f.x; p[1] = (__bf16)f.y; p[2] = (__bf16)f.z; p[3] = (__bf16)f.w;
  *(v4bf*)(xb + base) = p;
}

// ======================================================================
// Kernel 0b: w (f32 [K][N]) -> wb (bf16, WMMA B-layout chunks)
// chunk = 16 cols x 32 rows, 512 elems:
//   wb[((n>>4)*(K/32) + (k>>5))*512 + ((k>>4)&1)*256 + (n&15)*16 + (k&15)]
// Each thread converts 4 consecutive n of one k row (f32x4 -> 4 stores).
// ======================================================================
__global__ __launch_bounds__(256) void conv_w_kernel(
    const float* __restrict__ w, __bf16* __restrict__ wb, int K, int N) {
  const size_t base = ((size_t)blockIdx.x * 256 + threadIdx.x) * 4;
  const int k  = (int)(base / N);
  const int n0 = (int)(base % N);
  const float4 f = *(const float4*)(w + base);
  const float fv[4] = {f.x, f.y, f.z, f.w};
  const size_t rowOff = (size_t)(k >> 5) * 512 + (size_t)((k >> 4) & 1) * 256 + (k & 15);
  const size_t nStride = (size_t)(K >> 5) * 512;
#pragma unroll
  for (int j = 0; j < 4; ++j) {
    const int n = n0 + j;
    wb[(size_t)(n >> 4) * nStride + rowOff + (size_t)(n & 15) * 16] = (__bf16)fv[j];
  }
}

// ======================================================================
// Kernel 1: qkv = xb @ wbq + b_qkv (all-bf16 operands, LDS-free)
//   -> bf16 Q [bh][t][d], K [bh][t][d], V transposed [bh][d][t]
// block: 256 thr = 8 independent waves stacked on M. wave: 32M x 64N.
// ======================================================================
__global__ __launch_bounds__(256) void qkv_gemm_kernel(
    const __bf16* __restrict__ xb, const __bf16* __restrict__ wb,
    const float* __restrict__ bias, __bf16* __restrict__ qkv) {
  const int wave = threadIdx.x >> 5;
  const int lane = threadIdx.x & 31;
  const int hi   = lane >> 4;
  const int ln   = lane & 15;

  const int nWave = blockIdx.x * 64;
  const int mWave = blockIdx.y * 256 + wave * 32;
  const __bf16* arow0 = xb + (size_t)(mWave + ln) * C_;
  const __bf16* arow1 = xb + (size_t)(mWave + 16 + ln) * C_;
  // per-lane base inside each B chunk
  const __bf16* bbase = wb + (size_t)hi * 256 + (size_t)ln * 16;
  constexpr size_t KCH = C_ / 32;  // k-chunks per n-chunk

  v8f acc[2][4];
#pragma unroll
  for (int r = 0; r < 2; ++r)
#pragma unroll
    for (int nt = 0; nt < 4; ++nt)
#pragma unroll
      for (int i = 0; i < 8; ++i) acc[r][nt][i] = 0.0f;

  for (int kk = 0; kk < C_; kk += 32) {
    const size_t kc = (size_t)(kk >> 5);
    const v16bf a0 = load_a_bf16(arow0 + kk, hi);
    const v16bf a1 = load_a_bf16(arow1 + kk, hi);
#pragma unroll
    for (int nt = 0; nt < 4; ++nt) {
      const size_t nc = (size_t)((nWave >> 4) + nt);
      const v16bf b = *(const v16bf*)(bbase + (nc * KCH + kc) * 512);
      acc[0][nt] = WMMA_BF16(a0, b, acc[0][nt]);
      acc[1][nt] = WMMA_BF16(a1, b, acc[1][nt]);
    }
  }

  // ---- epilogue: bias, scatter bf16. Q/K -> [bh][t][d], V -> [bh][d][t] ----
#pragma unroll
  for (int nt = 0; nt < 4; ++nt) {
    const int n   = nWave + nt * 16 + ln;
    const int sel = n >> 10;
    const int rem = n & 1023;
    const int h   = rem >> 6;
    const int d   = rem & 63;
    const float bv = bias[n];
#pragma unroll
    for (int r = 0; r < 2; ++r)
#pragma unroll
      for (int i = 0; i < 8; ++i) {
        const int row = mWave + r * 16 + i + 8 * hi;
        const int bb  = row / T_;
        const int t   = row % T_;
        const float val = acc[r][nt][i] + bv;
        const size_t bh = (size_t)(bb * H_ + h);
        size_t idx;
        if (sel == 2) idx = 2 * BHTD + (bh * D_ + d) * T_ + t;   // V transposed
        else          idx = (size_t)sel * BHTD + (bh * T_ + t) * D_ + d;
        qkv[idx] = (__bf16)val;
      }
  }
}

// ======================================================================
// Kernel 2: flash attention. 1 wave = one 16-query tile of one (b,h).
// block: 128 thr = 4 independent waves.
// ======================================================================
__global__ __launch_bounds__(128) void flash_attn_kernel(
    const __bf16* __restrict__ qkv, __bf16* __restrict__ attb) {
  __shared__ __align__(16) __bf16 pl[4][16][40];  // per-wave P tile (padded pitch)

  const int wave = threadIdx.x >> 5;
  const int lane = threadIdx.x & 31;
  const int hi   = lane >> 4;
  const int ln   = lane & 15;

  const int nqt  = T_ / 16;                       // 128 query tiles per (b,h)
  const int tile = blockIdx.x * 4 + wave;
  const int bh   = tile / nqt;
  const int qt   = tile % nqt;
  const int q0   = qt * 16;

  const __bf16* Qh  = qkv + (size_t)bh * T_ * D_;
  const __bf16* Kh  = Qh + BHTD;
  const __bf16* VtH = qkv + 2 * BHTD + (size_t)bh * D_ * T_;  // [d][t]

  // ---- preload Q tile in A-layout (two K=32 chunks over d) ----
  const __bf16* qrow = Qh + (size_t)(q0 + ln) * D_;
  v16bf qa[2];
#pragma unroll
  for (int c = 0; c < 2; ++c) qa[c] = load_a_bf16(qrow + c * 32, hi);

  float m[8], l[8];
  v8f acc[4];
#pragma unroll
  for (int i = 0; i < 8; ++i) { m[i] = -1e30f; l[i] = 0.0f; }
#pragma unroll
  for (int nt = 0; nt < 4; ++nt)
#pragma unroll
    for (int i = 0; i < 8; ++i) acc[nt][i] = 0.0f;

  const float scale = 0.125f;                     // D^-0.5
  const int kEnd = q0 + 16;

  for (int k0 = 0; k0 < kEnd; k0 += 32) {
    // ---- S tiles: 16x16 each for keys [k0,k0+16) and [k0+16,k0+32) ----
    v8f s0, s1;
#pragma unroll
    for (int i = 0; i < 8; ++i) { s0[i] = 0.0f; s1[i] = 0.0f; }
    const int kr0 = imin(k0 + ln, T_ - 1);
    const int kr1 = imin(k0 + 16 + ln, T_ - 1);
#pragma unroll
    for (int c = 0; c < 2; ++c) {                 // K-dim (d) chunks of 32
      const v16bf kb0 = *(const v16bf*)(Kh + (size_t)kr0 * D_ + c * 32 + 16 * hi);
      const v16bf kb1 = *(const v16bf*)(Kh + (size_t)kr1 * D_ + c * 32 + 16 * hi);
      s0 = WMMA_BF16(qa[c], kb0, s0);
      s1 = WMMA_BF16(qa[c], kb1, s1);
    }

    __builtin_amdgcn_wave_barrier();
    // ---- scale, causal mask, online softmax, stash P in LDS ----
#pragma unroll
    for (int i = 0; i < 8; ++i) {
      const int row = q0 + i + 8 * hi;
      float a0 = s0[i] * scale;
      float a1 = s1[i] * scale;
      if (k0 + ln > row)      a0 = -1e30f;
      if (k0 + 16 + ln > row) a1 = -1e30f;

      float tmax = fmaxf(a0, a1);
      tmax = fmaxf(tmax, __shfl_xor(tmax, 1));
      tmax = fmaxf(tmax, __shfl_xor(tmax, 2));
      tmax = fmaxf(tmax, __shfl_xor(tmax, 4));
      tmax = fmaxf(tmax, __shfl_xor(tmax, 8));

      const float mn = fmaxf(m[i], tmax);
      const float f  = __expf(m[i] - mn);
      m[i] = mn;

      const float p0 = __expf(a0 - mn);
      const float p1 = __expf(a1 - mn);
      float rs = p0 + p1;
      rs += __shfl_xor(rs, 1);
      rs += __shfl_xor(rs, 2);
      rs += __shfl_xor(rs, 4);
      rs += __shfl_xor(rs, 8);
      l[i] = l[i] * f + rs;

      acc[0][i] *= f; acc[1][i] *= f; acc[2][i] *= f; acc[3][i] *= f;

      pl[wave][i + 8 * hi][ln]      = (__bf16)p0;
      pl[wave][i + 8 * hi][16 + ln] = (__bf16)p1;
    }
    __builtin_amdgcn_wave_barrier();

    // ---- reload P in A-layout (keys are the K=32 dim) ----
    const v16bf pa = load_a_bf16(&pl[wave][ln][0], hi);

    // ---- O += P @ V : Vt[d][t] gives contiguous B-layout loads ----
#pragma unroll
    for (int nt = 0; nt < 4; ++nt) {
      const v16bf vb =
          *(const v16bf*)(VtH + (size_t)(nt * 16 + ln) * T_ + k0 + 16 * hi);
      acc[nt] = WMMA_BF16(pa, vb, acc[nt]);
    }
  }

  // ---- normalize and write bf16 [B,T,C] ----
  const int bb = bh >> 4;
  const int h  = bh & 15;
#pragma unroll
  for (int i = 0; i < 8; ++i) {
    const float inv = 1.0f / l[i];
    const int row = q0 + i + 8 * hi;
#pragma unroll
    for (int nt = 0; nt < 4; ++nt) {
      attb[((size_t)(bb * T_ + row)) * C_ + h * D_ + nt * 16 + ln] =
          (__bf16)(acc[nt][i] * inv);
    }
  }
}

// ======================================================================
// Kernel 3: out = attb(bf16) @ wbo + b_out -> f32 d_out [B*T, C]
// LDS-free, same structure as kernel 1.
// ======================================================================
__global__ __launch_bounds__(256) void out_gemm_kernel(
    const __bf16* __restrict__ attb, const __bf16* __restrict__ wb,
    const float* __restrict__ bias, float* __restrict__ out) {
  const int wave = threadIdx.x >> 5;
  const int lane = threadIdx.x & 31;
  const int hi   = lane >> 4;
  const int ln   = lane & 15;

  const int nWave = blockIdx.x * 64;
  const int mWave = blockIdx.y * 256 + wave * 32;
  const __bf16* arow0 = attb + (size_t)(mWave + ln) * C_;
  const __bf16* arow1 = attb + (size_t)(mWave + 16 + ln) * C_;
  const __bf16* bbase = wb + (size_t)hi * 256 + (size_t)ln * 16;
  constexpr size_t KCH = C_ / 32;

  v8f acc[2][4];
#pragma unroll
  for (int r = 0; r < 2; ++r)
#pragma unroll
    for (int nt = 0; nt < 4; ++nt)
#pragma unroll
      for (int i = 0; i < 8; ++i) acc[r][nt][i] = 0.0f;

  for (int kk = 0; kk < C_; kk += 32) {
    const size_t kc = (size_t)(kk >> 5);
    const v16bf a0 = load_a_bf16(arow0 + kk, hi);
    const v16bf a1 = load_a_bf16(arow1 + kk, hi);
#pragma unroll
    for (int nt = 0; nt < 4; ++nt) {
      const size_t nc = (size_t)((nWave >> 4) + nt);
      const v16bf b = *(const v16bf*)(bbase + (nc * KCH + kc) * 512);
      acc[0][nt] = WMMA_BF16(a0, b, acc[0][nt]);
      acc[1][nt] = WMMA_BF16(a1, b, acc[1][nt]);
    }
  }

#pragma unroll
  for (int nt = 0; nt < 4; ++nt) {
    const int n = nWave + nt * 16 + ln;
    const float bv = bias[n];
#pragma unroll
    for (int r = 0; r < 2; ++r)
#pragma unroll
      for (int i = 0; i < 8; ++i) {
        const int row = mWave + r * 16 + i + 8 * hi;
        out[(size_t)row * C_ + n] = acc[r][nt][i] + bv;
      }
  }
}

// ======================================================================
extern "C" void kernel_launch(void* const* d_in, const int* in_sizes, int n_in,
                              void* d_out, int out_size, void* d_ws, size_t ws_size,
                              hipStream_t stream) {
  const float* x     = (const float*)d_in[0];
  const float* w_qkv = (const float*)d_in[1];
  const float* b_qkv = (const float*)d_in[2];
  const float* w_out = (const float*)d_in[3];
  const float* b_out = (const float*)d_in[4];
  float* out = (float*)d_out;

  // ---- workspace layout (bf16 elements) ----
  __bf16* qkv  = (__bf16*)d_ws;                   // Q,K [bh][t][d]; Vt [bh][d][t] : 3*BHTD
  __bf16* attb = qkv + 3 * BHTD;                  // B*T*C
  __bf16* xb   = attb + (size_t)M_ * C_;          // M*C
  __bf16* wbq  = xb + (size_t)M_ * C_;            // C*3C, B-layout chunks
  __bf16* wbo  = wbq + (size_t)C_ * N3_;          // C*C,  B-layout chunks

  // 0) precision/layout conversion (one-time per launch, L2-resident after)
  conv_x_kernel<<<dim3((M_ * C_) / 1024), 256, 0, stream>>>(x, xb);
  conv_w_kernel<<<dim3((C_ * N3_) / 1024), 256, 0, stream>>>(w_qkv, wbq, C_, N3_);
  conv_w_kernel<<<dim3((C_ * C_) / 1024), 256, 0, stream>>>(w_out, wbo, C_, C_);

  // 1) QKV projection: grid (3072/64, 8192/256)
  qkv_gemm_kernel<<<dim3(N3_ / 64, M_ / 256), 256, 0, stream>>>(xb, wbq, b_qkv, qkv);

  // 2) Flash attention: 8192 query tiles, 4 waves/block
  flash_attn_kernel<<<dim3(B_ * H_ * (T_ / 16) / 4), 128, 0, stream>>>(qkv, attb);

  // 3) Output projection: grid (1024/64, 8192/256)
  out_gemm_kernel<<<dim3(C_ / 64, M_ / 256), 256, 0, stream>>>(attb, wbo, b_out, out);
}
